// BoxAttention_4638564679872
// MI455X (gfx1250) — compile-verified, matrix operand order checked by reading, not run on hardware
//
#include <hip/hip_runtime.h>
#include <hip/hip_bf16.h>

// ---------------------------------------------------------------------------
// BoxAttention (Swin-style 3D window attention) for MI455X / gfx1250
//   x:(2048,64,384) f32 -> out:(2048,64,384) f32
//   Pipeline: f32->bf16 cvt, bias gather, QKV GEMM (WMMA bf16, TDM-staged,
//             double-buffered), fused per-(window,head) attention
//             (TDM-staged q/k/v/bias, WMMA bf16 + f32 softmax),
//             proj GEMM (WMMA bf16) + bias -> f32
// ---------------------------------------------------------------------------

typedef __attribute__((ext_vector_type(16))) __bf16 v16bf;
typedef __attribute__((ext_vector_type(8)))  float  v8f;
typedef __attribute__((ext_vector_type(4)))  unsigned int u32x4;
typedef __attribute__((ext_vector_type(8)))  int i32x8;
typedef __attribute__((ext_vector_type(4)))  int i32x4;

#define BBATCH 2048
#define NTOK   64
#define CDIM   384
#define NHEAD  12
#define HDIM   32
#define MROWS  (BBATCH * NTOK)   /* 131072 */
#define NQKV   (3 * CDIM)        /* 1152   */
#define QSCALE 0.17677669529663687f /* 32^-0.5 */

#if __has_builtin(__builtin_amdgcn_tensor_load_to_lds) && \
    __has_builtin(__builtin_amdgcn_s_wait_tensorcnt)
#define USE_TDM 1
#else
#define USE_TDM 0
#endif

// ---------------- TDM helpers (CDNA5 Tensor Data Mover) ---------------------

#if USE_TDM
// LDS byte offset of a __shared__ object: flat LDS aperture keeps the LDS
// address in addr[31:0] (ISA 10.2 aperture table).
__device__ __forceinline__ unsigned lds_off(const void* p) {
  return (unsigned)(unsigned long long)p;
}

// Issue a 2D tile DMA  global -> LDS  (one instruction per calling wave).
//   dsz_code: 0=1B, 1=2B, 2=4B elements
//   tile_w:   elements per row, tile_h: rows, row_stride: elements
// D# bit layout per CDNA5 ISA ch.8 (groups 2/3 zero => 2D tensor).
// This toolchain exposes the 6-arg builtin (g0, g1, g2, g3, g4, cpol).
__device__ __forceinline__ void tdm_load_2d(unsigned lds_addr, const void* gptr,
                                            unsigned dsz_code, unsigned tile_w,
                                            unsigned tile_h, unsigned row_stride) {
  unsigned long long ga = (unsigned long long)gptr;
  u32x4 g0;
  g0[0] = 1u;                                   // count=1, user mode, no gather
  g0[1] = lds_addr;                             // [63:32]  lds_addr
  g0[2] = (unsigned)(ga & 0xffffffffu);         // [95:64]  global_addr lo
  g0[3] = (unsigned)((ga >> 32) & 0x01ffffffu)  // [120:96] global_addr hi
          | (2u << 30);                         // [127:126] type=2 (image)
  const unsigned td0 = 0x40000000u, td1 = 0x40000000u;  // huge dims: no OOB clip
  i32x8 g1;
  g1[0] = (int)(dsz_code << 16);                        // mask=0, data_size
  g1[1] = (int)((td0 & 0xffffu) << 16);                 // [63:48] dim0 lo16
  g1[2] = (int)((td0 >> 16) | ((td1 & 0xffffu) << 16)); // dim0 hi16, dim1 lo16
  g1[3] = (int)((td1 >> 16) | (tile_w << 16));          // dim1 hi16, tile_dim0
  g1[4] = (int)(tile_h & 0xffffu);                      // tile_dim1, tile_dim2=0
  g1[5] = (int)row_stride;                              // dim0_stride lo32
  g1[6] = 0;                                            // stride hi, dim1_stride
  g1[7] = 0;
  i32x4 z4 = {0, 0, 0, 0};
  i32x8 z8 = {0, 0, 0, 0, 0, 0, 0, 0};
  __builtin_amdgcn_tensor_load_to_lds(g0, g1, z4, z4, z8, 0);
}
#endif

// ---------------- WMMA helpers (CDNA5 layouts, wave32) ----------------------

// K index for element e (0..15) of a 16-bit A/B fragment, lane-half h.
// Layout per ISA 7.12.2: VGPR0..3 hold K={0..7}+8h, VGPR4..7 hold K={16..23}+8h
__device__ __forceinline__ int frag_k(int e, int h) {
  int i = e >> 1, w2 = e & 1;
  int base = (i < 4) ? (2 * i) : (16 + 2 * (i - 4));
  return base + h * 8 + w2;
}

// Load a 16x32 bf16 fragment: element (r,k) at base[r*rs + k*cs].
// For A: r = M row. For B: r = N column (B stored as N x K).
__device__ __forceinline__ v16bf load_frag_bf(const __bf16* base, int rs, int cs) {
  int lane = threadIdx.x & 31;
  int r = lane & 15, h = lane >> 4;
  v16bf f;
#pragma unroll
  for (int e = 0; e < 16; ++e) f[e] = base[r * rs + frag_k(e, h) * cs];
  return f;
}

// Same, but source is f32 (softmax probabilities) converted to bf16 on load.
__device__ __forceinline__ v16bf load_frag_f32(const float* base, int rs, int cs) {
  int lane = threadIdx.x & 31;
  int r = lane & 15, h = lane >> 4;
  v16bf f;
#pragma unroll
  for (int e = 0; e < 16; ++e) f[e] = (__bf16)base[r * rs + frag_k(e, h) * cs];
  return f;
}

__device__ __forceinline__ v8f wmma_bf16(v16bf a, v16bf b, v8f c) {
  return __builtin_amdgcn_wmma_f32_16x16x32_bf16(
      /*neg_a=*/false, a, /*neg_b=*/false, b,
      /*c_mod=*/(short)0, c, /*reuse_a=*/false, /*reuse_b=*/false);
}

// ---------------- small utility kernels -------------------------------------

__global__ __launch_bounds__(256) void cvt_f32_bf16(const float* __restrict__ in,
                                                    __bf16* __restrict__ out, int n) {
  int i = blockIdx.x * blockDim.x + threadIdx.x;
  if (i < n) out[i] = (__bf16)in[i];
}

// bias[h][i][j] = bias_table[rel_idx[i*64+j] * H + h]
__global__ __launch_bounds__(256) void build_bias(const float* __restrict__ table,
                                                  const int* __restrict__ rel_idx,
                                                  float* __restrict__ bias) {
  int t = blockIdx.x * blockDim.x + threadIdx.x;
  if (t < NHEAD * NTOK * NTOK) {
    int h = t / (NTOK * NTOK);
    int ij = t % (NTOK * NTOK);
    bias[t] = table[rel_idx[ij] * NHEAD + h];
  }
}

// ---------------- tiled WMMA GEMM: out[M,N] = A[M,K] * Bw[N,K]^T ------------
// 256 threads = 8 waves; block tile 128(M) x 128(N); K-step 32.
// Wave (wm 0..3, wn 0..1) owns a 32x64 sub-tile = 2x4 accumulator frags.
// TDM path: wave 0 DMAs the next K tile into the alternate LDS buffer while
// all 8 waves run WMMA on the current one (double-buffered, TENSORcnt-synced).
template <int NDIM, bool SCALE_Q, bool ADD_PB, bool OUT_BF16>
__global__ __launch_bounds__(256) void gemm_bf16_kernel(
    const __bf16* __restrict__ A, const __bf16* __restrict__ Bw,
    const float* __restrict__ pbias, __bf16* __restrict__ out_bf,
    float* __restrict__ out_f32) {
  constexpr int KDIM = 384;
  constexpr int KSTEPS = KDIM / 32;
  __shared__ __bf16 As[2][128 * 32];
  __shared__ __bf16 Bs[2][128 * 32];
  const int bm = blockIdx.x * 128;
  const int bn = blockIdx.y * 128;
  const int wid = threadIdx.x >> 5;
  const int wm = wid & 3, wn = wid >> 2;
  v8f acc[2][4] = {};

#if USE_TDM
  if (wid == 0) {
    tdm_load_2d(lds_off(&As[0][0]), A + (size_t)bm * KDIM, 1, 32, 128, KDIM);
    tdm_load_2d(lds_off(&Bs[0][0]), Bw + (size_t)bn * KDIM, 1, 32, 128, KDIM);
    __builtin_amdgcn_s_wait_tensorcnt(0);
  }
  __syncthreads();
#endif

  for (int it = 0; it < KSTEPS; ++it) {
    const int cur = it & 1;
#if USE_TDM
    const int nxt = cur ^ 1;
    const int k0n = (it + 1) * 32;
    if (k0n < KDIM && wid == 0) {  // prefetch next tile while computing
      tdm_load_2d(lds_off(&As[nxt][0]), A + (size_t)bm * KDIM + k0n, 1, 32, 128, KDIM);
      tdm_load_2d(lds_off(&Bs[nxt][0]), Bw + (size_t)bn * KDIM + k0n, 1, 32, 128, KDIM);
    }
#else
    const int k0 = it * 32;
    __syncthreads();
    {
      const uint4* A4 = (const uint4*)A;
      const uint4* B4 = (const uint4*)Bw;
#pragma unroll
      for (int i = 0; i < 2; ++i) {
        int lin = threadIdx.x + i * 256;   // 512 uint4 per tile
        int row = lin >> 2, c4 = lin & 3;  // 4 uint4 (32 bf16) per row
        ((uint4*)As[cur])[lin] = A4[(size_t)(bm + row) * (KDIM / 8) + (k0 >> 3) + c4];
        ((uint4*)Bs[cur])[lin] = B4[(size_t)(bn + row) * (KDIM / 8) + (k0 >> 3) + c4];
      }
    }
    __syncthreads();
#endif
    v16bf a0 = load_frag_bf(As[cur] + (wm * 32 + 0) * 32, 32, 1);
    v16bf a1 = load_frag_bf(As[cur] + (wm * 32 + 16) * 32, 32, 1);
#pragma unroll
    for (int j = 0; j < 4; ++j) {
      v16bf b = load_frag_bf(Bs[cur] + (wn * 64 + j * 16) * 32, 32, 1);
      acc[0][j] = wmma_bf16(a0, b, acc[0][j]);
      acc[1][j] = wmma_bf16(a1, b, acc[1][j]);
    }
#if USE_TDM
    __syncthreads();  // all waves done reading buffer 'cur'
    if (k0n < KDIM) {
      if (wid == 0) __builtin_amdgcn_s_wait_tensorcnt(0);
      __syncthreads();  // next buffer visible to every wave
    }
#endif
  }

  // Epilogue: C/D layout — VGPR r: lanes 0-15 => M=r, lanes 16-31 => M=r+8.
  const int lane = threadIdx.x & 31;
  const int nn = lane & 15, mhalf = (lane >> 4) * 8;
#pragma unroll
  for (int mi = 0; mi < 2; ++mi)
#pragma unroll
    for (int j = 0; j < 4; ++j)
#pragma unroll
      for (int r = 0; r < 8; ++r) {
        int gm = bm + wm * 32 + mi * 16 + mhalf + r;
        int gn = bn + wn * 64 + j * 16 + nn;
        float v = acc[mi][j][r];
        if (SCALE_Q && gn < CDIM) v *= QSCALE;  // pre-scale q columns
        if (ADD_PB) v += pbias[gn];
        if (OUT_BF16)
          out_bf[(size_t)gm * NDIM + gn] = (__bf16)v;
        else
          out_f32[(size_t)gm * NDIM + gn] = v;
      }
}

// ---------------- fused attention: one block per (window, head) -------------
// 128 threads = 4 waves. Wave w owns token rows [16w, 16w+16).
//   S = q k^T (+bias) -> softmax (f32, LDS) -> O = P v
// q/k/v head slices (64x32 bf16, stride 1152) and the bias tile (64x64 f32)
// are staged by four TDM descriptors issued from wave 0.
__global__ __launch_bounds__(128) void attn_kernel(
    const __bf16* __restrict__ qkv, const float* __restrict__ bias,
    __bf16* __restrict__ ao) {
  __shared__ __bf16 qs[NTOK * HDIM];
  __shared__ __bf16 ks[NTOK * HDIM];
  __shared__ __bf16 vs[NTOK * HDIM];
  __shared__ float  bls[NTOK * NTOK];
  __shared__ float  Ss[NTOK * NTOK];

  const int h = blockIdx.x % NHEAD;
  const int b = blockIdx.x / NHEAD;

#if USE_TDM
  if ((threadIdx.x >> 5) == 0) {
    const __bf16* base = qkv + (size_t)b * NTOK * NQKV + h * HDIM;
    tdm_load_2d(lds_off(qs), base,            1, HDIM, NTOK, NQKV);
    tdm_load_2d(lds_off(ks), base + CDIM,     1, HDIM, NTOK, NQKV);
    tdm_load_2d(lds_off(vs), base + 2 * CDIM, 1, HDIM, NTOK, NQKV);
    tdm_load_2d(lds_off(bls), bias + (size_t)h * NTOK * NTOK, 2, NTOK, NTOK, NTOK);
    __builtin_amdgcn_s_wait_tensorcnt(0);
  }
  __syncthreads();
#else
  {
    const uint4* q4 = (const uint4*)qkv;
#pragma unroll
    for (int i = 0; i < 2; ++i) {
      int lin = threadIdx.x + i * 128;  // 256 uint4 per matrix
      int row = lin >> 2, c4 = lin & 3;
      size_t base = (size_t)(b * NTOK + row) * (NQKV / 8) + h * 4 + c4;
      ((uint4*)qs)[lin] = q4[base];
      ((uint4*)ks)[lin] = q4[base + CDIM / 8];
      ((uint4*)vs)[lin] = q4[base + 2 * CDIM / 8];
    }
    const float* bh = bias + (size_t)h * NTOK * NTOK;
#pragma unroll
    for (int i = 0; i < NTOK * NTOK / 128; ++i)
      bls[threadIdx.x + i * 128] = bh[threadIdx.x + i * 128];
  }
  __syncthreads();
#endif

  const int w = threadIdx.x >> 5;
  const int lane = threadIdx.x & 31;
  const int nn = lane & 15, mhalf = (lane >> 4) * 8;

  // S[16w..16w+15][0..63] = q_rows @ k^T  (B frag: N = k-token, K = hd)
  {
    v16bf aq = load_frag_bf(qs + (w * 16) * HDIM, HDIM, 1);
    v8f s[4] = {};
#pragma unroll
    for (int j = 0; j < 4; ++j) {
      v16bf bk = load_frag_bf(ks + (j * 16) * HDIM, HDIM, 1);
      s[j] = wmma_bf16(aq, bk, s[j]);
    }
#pragma unroll
    for (int j = 0; j < 4; ++j)
#pragma unroll
      for (int r = 0; r < 8; ++r) {
        int m = w * 16 + mhalf + r, n = j * 16 + nn;
        Ss[m * NTOK + n] = s[j][r] + bls[m * NTOK + n];
      }
  }
  __syncthreads();

  // Row-wise softmax in f32 (one thread per row).
  if (threadIdx.x < NTOK) {
    float* row = Ss + threadIdx.x * NTOK;
    float mx = row[0];
#pragma unroll
    for (int i = 1; i < NTOK; ++i) mx = fmaxf(mx, row[i]);
    float sum = 0.f;
#pragma unroll
    for (int i = 0; i < NTOK; ++i) { float e = __expf(row[i] - mx); row[i] = e; sum += e; }
    float inv = 1.f / sum;
#pragma unroll
    for (int i = 0; i < NTOK; ++i) row[i] *= inv;
  }
  __syncthreads();

  // O[16w..][0..31] = P @ v.  A = P rows (f32->bf16), two K blocks of 32.
  // B frag for (kb, nb): element (n=d_local, k=token_local) at
  //   vs[(kb*32 + k)*32 + nb*16 + n]  -> base + n*1 + k*32.
  v8f o[2] = {};
#pragma unroll
  for (int kb = 0; kb < 2; ++kb) {
    v16bf ap = load_frag_f32(Ss + (w * 16) * NTOK + kb * 32, NTOK, 1);
#pragma unroll
    for (int nb = 0; nb < 2; ++nb) {
      v16bf bv = load_frag_bf(vs + kb * 32 * HDIM + nb * 16, 1, HDIM);
      o[nb] = wmma_bf16(ap, bv, o[nb]);
    }
  }
#pragma unroll
  for (int nb = 0; nb < 2; ++nb)
#pragma unroll
    for (int r = 0; r < 8; ++r) {
      int m = w * 16 + mhalf + r, d = nb * 16 + nn;
      ao[((size_t)b * NTOK + m) * CDIM + h * HDIM + d] = (__bf16)o[nb][r];
    }
}

// ---------------------------------------------------------------------------

extern "C" void kernel_launch(void* const* d_in, const int* in_sizes, int n_in,
                              void* d_out, int out_size, void* d_ws, size_t ws_size,
                              hipStream_t stream) {
  const float* x          = (const float*)d_in[0];  // (2048,64,384)
  const float* qkv_w      = (const float*)d_in[1];  // (1152,384)
  const float* proj_w     = (const float*)d_in[2];  // (384,384)
  const float* proj_b     = (const float*)d_in[3];  // (384,)
  const float* bias_table = (const float*)d_in[4];  // (121,12)
  const int*   rel_idx    = (const int*)d_in[5];    // (64,64)
  float* out = (float*)d_out;

  // Workspace carve-up (256B aligned)
  size_t off = 0;
  auto carve = [&](size_t bytes) -> void* {
    void* p = (char*)d_ws + off;
    off += (bytes + 255) & ~(size_t)255;
    return p;
  };
  __bf16* x_bf   = (__bf16*)carve((size_t)MROWS * CDIM * 2);       // 100.7 MB
  __bf16* wq_bf  = (__bf16*)carve((size_t)NQKV * CDIM * 2);        // 0.9 MB
  __bf16* wp_bf  = (__bf16*)carve((size_t)CDIM * CDIM * 2);        // 0.3 MB
  __bf16* qkv_bf = (__bf16*)carve((size_t)MROWS * NQKV * 2);       // 302 MB
  float*  bias_f = (float*)carve((size_t)NHEAD * NTOK * NTOK * 4); // 0.2 MB
  __bf16* ao_bf  = (__bf16*)carve((size_t)MROWS * CDIM * 2);       // 100.7 MB
  (void)ws_size; (void)in_sizes; (void)n_in; (void)out_size;

  // 1) precision conversion + bias gather
  {
    int n = MROWS * CDIM;
    cvt_f32_bf16<<<(n + 255) / 256, 256, 0, stream>>>(x, x_bf, n);
    n = NQKV * CDIM;
    cvt_f32_bf16<<<(n + 255) / 256, 256, 0, stream>>>(qkv_w, wq_bf, n);
    n = CDIM * CDIM;
    cvt_f32_bf16<<<(n + 255) / 256, 256, 0, stream>>>(proj_w, wp_bf, n);
    n = NHEAD * NTOK * NTOK;
    build_bias<<<(n + 255) / 256, 256, 0, stream>>>(bias_table, rel_idx, bias_f);
  }

  // 2) QKV GEMM: qkv_bf[131072,1152] = x_bf @ wq_bf^T (q cols pre-scaled)
  gemm_bf16_kernel<NQKV, true, false, true>
      <<<dim3(MROWS / 128, NQKV / 128), 256, 0, stream>>>(
          x_bf, wq_bf, nullptr, qkv_bf, nullptr);

  // 3) fused attention per (window, head)
  attn_kernel<<<BBATCH * NHEAD, 128, 0, stream>>>(qkv_bf, bias_f, ao_bf);

  // 4) proj GEMM + bias -> f32 out
  gemm_bf16_kernel<CDIM, false, true, false>
      <<<dim3(MROWS / 128, CDIM / 128), 256, 0, stream>>>(
          ao_bf, wp_bf, proj_b, nullptr, out);
}
